// LSTM_B_45741401702887
// MI455X (gfx1250) — matrix operand order repaired and checked by reading, not compiled, not run on hardware
//
#include <hip/hip_runtime.h>

typedef __attribute__((ext_vector_type(16))) _Float16 v16h;
typedef __attribute__((ext_vector_type(8)))  _Float16 v8h;
typedef __attribute__((ext_vector_type(4)))  _Float16 v4h;
typedef __attribute__((ext_vector_type(8)))  float    v8f;

constexpr int kB   = 256;   // batch
constexpr int kH   = 50;    // hidden
constexpr int kNG  = 208;   // 4H=200 padded to 13 WMMA n-tiles
constexpr int kNT  = 13;    // gate n-tiles
constexpr int kKL0 = 64;    // layer0 fused K: 4(x)+50(h) padded
constexpr int kKH  = 64;    // recurrent K: 50 padded
constexpr int kKI1 = 128;   // layer1 input-proj K: 100 padded
constexpr int kH1S = 128;   // h1 row stride (halves), zero-padded 100..127
constexpr int kXPS = 256;   // xp LDS staging row stride (floats)

#if defined(__AMDGCN__) && __has_builtin(__builtin_amdgcn_global_load_async_to_lds_b128) && \
    __has_builtin(__builtin_amdgcn_s_wait_asynccnt)
#define HAS_ASYNC 1
#else
#define HAS_ASYNC 0
#endif

#if HAS_ASYNC
#define AS1 __attribute__((address_space(1)))
#define AS3 __attribute__((address_space(3)))
typedef __attribute__((ext_vector_type(4))) int v4i;
__device__ __forceinline__ void async_copy_b128(const void* g, void* l) {
  __builtin_amdgcn_global_load_async_to_lds_b128(
      (AS1 v4i*)(uintptr_t)g, (AS3 v4i*)(uint32_t)(uintptr_t)l, 0, 0);
}
#endif

__device__ __forceinline__ v8f wmma16(v16h a, v16h b, v8f c) {
  return __builtin_amdgcn_wmma_f32_16x16x32_f16(false, a, false, b, (short)0, c,
                                                false, false);
}

// A fragment (16xK row-major, element stride `strideH` halves) for v_wmma_*_16x16x32
__device__ __forceinline__ v16h load_a_frag(const _Float16* A, size_t strideH,
                                            int k0, int lane) {
  const int row = lane & 15;
  const int kb  = (lane >> 4) * 8;
  union { v16h v; v8h h[2]; } u;
  u.h[0] = *(const v8h*)(A + (size_t)row * strideH + (k0 + kb));
  u.h[1] = *(const v8h*)(A + (size_t)row * strideH + (k0 + 16 + kb));
  return u.v;
}

// B fragment (KxN stored n-major [n][k])
__device__ __forceinline__ v16h load_b_frag(const _Float16* B, size_t strideH,
                                            int n0, int k0, int lane) {
  const int col = n0 + (lane & 15);
  const int kb  = (lane >> 4) * 8;
  union { v16h v; v8h h[2]; } u;
  u.h[0] = *(const v8h*)(B + (size_t)col * strideH + (k0 + kb));
  u.h[1] = *(const v8h*)(B + (size_t)col * strideH + (k0 + 16 + kb));
  return u.v;
}

__device__ __forceinline__ float sigm(float x) { return 1.f / (1.f + __expf(-x)); }
__device__ __forceinline__ float tanh_(float x) {
  const float e = __expf(-2.f * x);
  return (1.f - e) / (1.f + e);
}

// ---------------- weight packing (f32 -> f16 WMMA-B layouts) ----------------
__global__ void pack_weights(const float* __restrict__ W0f_ih, const float* __restrict__ W0f_hh,
                             const float* __restrict__ W0b_ih, const float* __restrict__ W0b_hh,
                             const float* __restrict__ W1f_hh, const float* __restrict__ W1f_ih,
                             const float* __restrict__ W1b_ih,
                             _Float16* __restrict__ B0, _Float16* __restrict__ B1hh,
                             _Float16* __restrict__ B1if, _Float16* __restrict__ B1ib) {
  int idx = blockIdx.x * blockDim.x + threadIdx.x;
  const int n0 = 2 * kNG * kKL0;                // stacked [Wih;Whh], both dirs
  if (idx < n0) {
    const int dir = idx / (kNG * kKL0);
    const int rem = idx - dir * (kNG * kKL0);
    const int n = rem / kKL0, k = rem - n * kKL0;
    const float* Wih = dir ? W0b_ih : W0f_ih;
    const float* Whh = dir ? W0b_hh : W0f_hh;
    float v = 0.f;
    if (n < 200) {
      if (k < 4)       v = Wih[n * 4 + k];
      else if (k < 54) v = Whh[n * 50 + (k - 4)];
    }
    B0[idx] = (_Float16)v;
    return;
  }
  idx -= n0;
  const int n1 = kNG * kKH;
  if (idx < n1) {
    const int n = idx / kKH, k = idx - n * kKH;
    B1hh[idx] = (_Float16)((n < 200 && k < 50) ? W1f_hh[n * 50 + k] : 0.f);
    return;
  }
  idx -= n1;
  const int n2 = kNG * kKI1;
  if (idx < n2) {
    const int n = idx / kKI1, k = idx - n * kKI1;
    B1if[idx] = (_Float16)((n < 200 && k < 100) ? W1f_ih[n * 100 + k] : 0.f);
    return;
  }
  idx -= n2;
  if (idx < n2) {
    const int n = idx / kKI1, k = idx - n * kKI1;
    B1ib[idx] = (_Float16)((n < 200 && k < 100) ? W1b_ih[n * 100 + k] : 0.f);
  }
}

__global__ void zero_h1_pad(_Float16* __restrict__ h1, size_t rows) {
  const size_t r = (size_t)blockIdx.x * blockDim.x + threadIdx.x;
  if (r < rows) {
    _Float16* p = h1 + r * kH1S + 100;   // pad cols 100..127
    const v4h z = {};
#pragma unroll
    for (int i = 0; i < 7; ++i) ((v4h*)p)[i] = z;
  }
}

// ---------------- layer 0: fused-input bi-LSTM scan ----------------
// grid.x = 32 : dir = blockIdx.x>>4, batch tile = blockIdx.x&15
__global__ __launch_bounds__(256) void layer0_scan(
    const float* __restrict__ seq,       // (B,4,T)
    const _Float16* __restrict__ B0,     // [2][kNG][kKL0]
    const float* __restrict__ b0f, const float* __restrict__ b0b,
    _Float16* __restrict__ h1,           // (B,T,kH1S) f16
    int T) {
  const int dir = blockIdx.x >> 4;
  const int b0  = (blockIdx.x & 15) * 16;
  const _Float16* Bm = B0 + (size_t)dir * kNG * kKL0;
  const float* bias  = dir ? b0b : b0f;

  __shared__ __align__(32) _Float16 Ast[16 * kKL0];  // [x_t | h | pad]
  __shared__ float gates[16 * kNG];
  __shared__ float cst[16 * kKH];

  const int tid = threadIdx.x, wave = tid >> 5, lane = tid & 31;
  for (int i = tid; i < 16 * kKL0; i += 256) Ast[i] = (_Float16)0.f;
  for (int i = tid; i < 16 * kKH; i += 256)  cst[i] = 0.f;

  const int nt0 = wave, nt1 = wave + 8;
  const int nt1c = (nt1 < kNT) ? nt1 : 0;
  const v16h bf00 = load_b_frag(Bm, kKL0, nt0 * 16, 0, lane);
  const v16h bf01 = load_b_frag(Bm, kKL0, nt0 * 16, 32, lane);
  const v16h bf10 = load_b_frag(Bm, kKL0, nt1c * 16, 0, lane);
  const v16h bf11 = load_b_frag(Bm, kKL0, nt1c * 16, 32, lane);

  const int colL = lane & 15, rbase = (lane >> 4) * 8;
  const int col0 = nt0 * 16 + colL, col1 = nt1 * 16 + colL;
  const float bv0 = (col0 < 200) ? bias[col0] : 0.f;
  const float bv1 = (nt1 < kNT && col1 < 200) ? bias[col1] : 0.f;
  const int rA = tid >> 4, j0 = tid & 15;          // activation mapping
  const int rs = tid >> 2, ch = tid & 3;           // x staging mapping

  // register double-buffer for x_t: load t=0 now, always load t+1 during step t
  float xreg = 0.f;
  if (tid < 64)
    xreg = seq[((size_t)(b0 + rs) * 4 + ch) * T + (dir ? (T - 1) : 0)];

  for (int step = 0; step < T; ++step) {
    const int t = dir ? (T - 1 - step) : step;
    if (tid < 64) {
      Ast[rs * kKL0 + ch] = (_Float16)xreg;        // x for this step
      if (step + 1 < T) {                          // overlap next load with step
        const int t2 = dir ? (T - 2 - step) : (step + 1);
        xreg = seq[((size_t)(b0 + rs) * 4 + ch) * T + t2];
      }
    }
    __syncthreads();

    const v16h a0 = load_a_frag(Ast, kKL0, 0, lane);
    const v16h a1 = load_a_frag(Ast, kKL0, 32, lane);
    {
      v8f acc = {};
      acc = wmma16(a0, bf00, acc);
      acc = wmma16(a1, bf01, acc);
#pragma unroll
      for (int r = 0; r < 8; ++r) gates[(rbase + r) * kNG + col0] = acc[r] + bv0;
    }
    if (nt1 < kNT) {
      v8f acc = {};
      acc = wmma16(a0, bf10, acc);
      acc = wmma16(a1, bf11, acc);
#pragma unroll
      for (int r = 0; r < 8; ++r) gates[(rbase + r) * kNG + col1] = acc[r] + bv1;
    }
    __syncthreads();

    // gate order i,f,g,o ; division-free indexing
#pragma unroll
    for (int pass = 0; pass < 4; ++pass) {
      const int j = j0 + pass * 16;
      if (j < kH) {
        const float gi = gates[rA * kNG + j];
        const float gf = gates[rA * kNG + 50 + j];
        const float gg = gates[rA * kNG + 100 + j];
        const float go = gates[rA * kNG + 150 + j];
        const float c  = sigm(gf) * cst[rA * kKH + j] + sigm(gi) * tanh_(gg);
        cst[rA * kKH + j] = c;
        const float h = sigm(go) * tanh_(c);
        Ast[rA * kKL0 + 4 + j] = (_Float16)h;      // feed next step
        h1[((size_t)(b0 + rA) * T + t) * kH1S + dir * kH + j] = (_Float16)h;
      }
    }
    // next iteration's barrier separates these writes from the GEMM reads
  }
}

// ---------------- layer 1 forward: big input-projection GEMM ----------------
__global__ __launch_bounds__(256) void xp1_gemm(
    const _Float16* __restrict__ h1, const _Float16* __restrict__ B1if,
    const float* __restrict__ b1f, float* __restrict__ xp1, int Mtiles) {
  const int wave = threadIdx.x >> 5, lane = threadIdx.x & 31;
  const int mt = blockIdx.x * 8 + wave;
  if (mt >= Mtiles) return;
  const _Float16* Arow = h1 + (size_t)mt * 16 * kH1S;
  v16h a[4];
#pragma unroll
  for (int s = 0; s < 4; ++s) a[s] = load_a_frag(Arow, kH1S, s * 32, lane);
  const int colL = lane & 15, rbase = (lane >> 4) * 8;
  float* out = xp1 + (size_t)mt * 16 * kNG;
  for (int nt = 0; nt < kNT; ++nt) {
    v8f acc = {};
#pragma unroll
    for (int s = 0; s < 4; ++s)
      acc = wmma16(a[s], load_b_frag(B1if, kKI1, nt * 16, s * 32, lane), acc);
    const int col = nt * 16 + colL;
    const float bv = (col < 200) ? b1f[col] : 0.f;
#pragma unroll
    for (int r = 0; r < 8; ++r) out[(size_t)(rbase + r) * kNG + col] = acc[r] + bv;
  }
}

// ---------------- layer 1 forward: recurrent scan (K=64 only) ----------------
__global__ __launch_bounds__(256) void layer1f_scan(
    const _Float16* __restrict__ B1hh,   // [kNG][kKH]
    const float* __restrict__ xp1,       // (B,T,kNG) f32
    float* __restrict__ h2f,             // (B,kH) final h
    int T) {
  const int b0 = blockIdx.x * 16;
  __shared__ __align__(32) _Float16 Ast[16 * kKH];
  __shared__ float gates[16 * kNG];
  __shared__ float cst[16 * kKH];
#if HAS_ASYNC
  __shared__ float xps[2][16 * kXPS];    // double-buffered async gate tiles
#endif
  const int tid = threadIdx.x, wave = tid >> 5, lane = tid & 31;
  for (int i = tid; i < 16 * kKH; i += 256) { Ast[i] = (_Float16)0.f; cst[i] = 0.f; }

  const int nt0 = wave, nt1 = wave + 8;
  const int nt1c = (nt1 < kNT) ? nt1 : 0;
  const v16h bf00 = load_b_frag(B1hh, kKH, nt0 * 16, 0, lane);
  const v16h bf01 = load_b_frag(B1hh, kKH, nt0 * 16, 32, lane);
  const v16h bf10 = load_b_frag(B1hh, kKH, nt1c * 16, 0, lane);
  const v16h bf11 = load_b_frag(B1hh, kKH, nt1c * 16, 32, lane);
  const int colL = lane & 15, rbase = (lane >> 4) * 8;
  const int col0 = nt0 * 16 + colL, col1 = nt1 * 16 + colL;
  const int rA = tid >> 4, j0 = tid & 15;

#if HAS_ASYNC
  // each thread copies exactly 4x b128: row rA, 16B chunks j0, j0+16, j0+32, j0+48
  {
    const float* gsrc = xp1 + ((size_t)(b0 + rA) * T + 0) * kNG;
    float* ldst = &xps[0][rA * kXPS];
#pragma unroll
    for (int k = 0; k < 4; ++k)
      async_copy_b128(gsrc + (j0 + 16 * k) * 4, ldst + (j0 + 16 * k) * 4);
  }
#endif
  __syncthreads();

  for (int t = 0; t < T; ++t) {
#if HAS_ASYNC
    const int p = t & 1;
    if (t + 1 < T) {                     // prefetch next gate tile into other buf
      const float* gsrc = xp1 + ((size_t)(b0 + rA) * T + (t + 1)) * kNG;
      float* ldst = &xps[p ^ 1][rA * kXPS];
#pragma unroll
      for (int k = 0; k < 4; ++k)
        async_copy_b128(gsrc + (j0 + 16 * k) * 4, ldst + (j0 + 16 * k) * 4);
      __builtin_amdgcn_s_wait_asynccnt(4);   // in-order: this step's tile landed
    } else {
      __builtin_amdgcn_s_wait_asynccnt(0);
    }
    __syncthreads();                     // tile visible to all waves; Ast ready
#else
    if (t + 1 < T && tid < 112) {
      const int r = tid & 15, s = tid >> 4;
      __builtin_prefetch(xp1 + ((size_t)(b0 + r) * T + (t + 1)) * kNG + s * 32, 0, 0);
    }
#endif
    const v16h a0 = load_a_frag(Ast, kKH, 0, lane);
    const v16h a1 = load_a_frag(Ast, kKH, 32, lane);
    {
      v8f acc = {};
      acc = wmma16(a0, bf00, acc);
      acc = wmma16(a1, bf01, acc);
#pragma unroll
      for (int r = 0; r < 8; ++r) {
#if HAS_ASYNC
        const float xpv = xps[p][(rbase + r) * kXPS + col0];
#else
        const float xpv = xp1[((size_t)(b0 + rbase + r) * T + t) * kNG + col0];
#endif
        gates[(rbase + r) * kNG + col0] = acc[r] + xpv;
      }
    }
    if (nt1 < kNT) {
      v8f acc = {};
      acc = wmma16(a0, bf10, acc);
      acc = wmma16(a1, bf11, acc);
#pragma unroll
      for (int r = 0; r < 8; ++r) {
#if HAS_ASYNC
        const float xpv = xps[p][(rbase + r) * kXPS + col1];
#else
        const float xpv = xp1[((size_t)(b0 + rbase + r) * T + t) * kNG + col1];
#endif
        gates[(rbase + r) * kNG + col1] = acc[r] + xpv;
      }
    }
    __syncthreads();
#pragma unroll
    for (int pass = 0; pass < 4; ++pass) {
      const int j = j0 + pass * 16;
      if (j < kH) {
        const float gi = gates[rA * kNG + j];
        const float gf = gates[rA * kNG + 50 + j];
        const float gg = gates[rA * kNG + 100 + j];
        const float go = gates[rA * kNG + 150 + j];
        const float c  = sigm(gf) * cst[rA * kKH + j] + sigm(gi) * tanh_(gg);
        cst[rA * kKH + j] = c;
        const float h = sigm(go) * tanh_(c);
        Ast[rA * kKH + j] = (_Float16)h;
        if (t == T - 1) h2f[(b0 + rA) * kH + j] = h;   // only last h consumed
      }
    }
#if !HAS_ASYNC
    __syncthreads();
#endif
  }
}

// ---------------- layer 1 backward: single step (h0=c0=0) ----------------
__global__ __launch_bounds__(256) void layer1b_step(
    const _Float16* __restrict__ h1, const _Float16* __restrict__ B1ib,
    const float* __restrict__ b1b, float* __restrict__ h2b, int T) {
  const int b0 = blockIdx.x * 16;
  __shared__ float gates[16 * kNG];
  const int tid = threadIdx.x, wave = tid >> 5, lane = tid & 31;
  const _Float16* Abase = h1 + ((size_t)b0 * T + (T - 1)) * kH1S;  // rows at t=T-1
  const size_t astride = (size_t)T * kH1S;
  v16h a[4];
#pragma unroll
  for (int s = 0; s < 4; ++s) a[s] = load_a_frag(Abase, astride, s * 32, lane);
  const int colL = lane & 15, rbase = (lane >> 4) * 8;
  for (int u = 0; u < 2; ++u) {
    const int nt = u ? (wave + 8) : wave;
    if (nt >= kNT) break;
    v8f acc = {};
#pragma unroll
    for (int s = 0; s < 4; ++s)
      acc = wmma16(a[s], load_b_frag(B1ib, kKI1, nt * 16, s * 32, lane), acc);
    const int col = nt * 16 + colL;
    const float bv = (col < 200) ? b1b[col] : 0.f;
#pragma unroll
    for (int r = 0; r < 8; ++r) gates[(rbase + r) * kNG + col] = acc[r] + bv;
  }
  __syncthreads();
  const int rA = tid >> 4, j0 = tid & 15;
#pragma unroll
  for (int pass = 0; pass < 4; ++pass) {
    const int j = j0 + pass * 16;
    if (j < kH) {
      const float gi = gates[rA * kNG + j];
      const float gg = gates[rA * kNG + 100 + j];
      const float go = gates[rA * kNG + 150 + j];
      const float c = sigm(gi) * tanh_(gg);          // f*c0 = 0
      h2b[(b0 + rA) * kH + j] = sigm(go) * tanh_(c);
    }
  }
}

// ---------------- head: linear + log_softmax, writes (out, logprob) ----------------
__global__ void head_k(const float* __restrict__ h2f, const float* __restrict__ h2b,
                       const float* __restrict__ Wl, const float* __restrict__ bl,
                       float* __restrict__ out) {
  const int b = blockIdx.x * blockDim.x + threadIdx.x;
  if (b >= kB) return;
  float acc[12];
#pragma unroll
  for (int c = 0; c < 12; ++c) acc[c] = bl[c];
  for (int j = 0; j < kH; ++j) {
    const float hf = h2f[b * kH + j], hb = h2b[b * kH + j];
#pragma unroll
    for (int c = 0; c < 12; ++c)
      acc[c] += hf * Wl[c * 100 + j] + hb * Wl[c * 100 + 50 + j];
  }
  float mx = acc[0];
#pragma unroll
  for (int c = 1; c < 12; ++c) mx = fmaxf(mx, acc[c]);
  float se = 0.f;
#pragma unroll
  for (int c = 0; c < 12; ++c) se += __expf(acc[c] - mx);
  const float lse = mx + __logf(se);
#pragma unroll
  for (int c = 0; c < 12; ++c) {
    out[b * 12 + c] = acc[c];                    // tuple[0]
    out[kB * 12 + b * 12 + c] = acc[c] - lse;    // tuple[1] log_softmax
  }
}

extern "C" void kernel_launch(void* const* d_in, const int* in_sizes, int n_in,
                              void* d_out, int out_size, void* d_ws, size_t ws_size,
                              hipStream_t stream) {
  const float* seq    = (const float*)d_in[0];
  const float* W0f_ih = (const float*)d_in[1];
  const float* W0f_hh = (const float*)d_in[2];
  const float* b0f    = (const float*)d_in[3];
  const float* W0b_ih = (const float*)d_in[4];
  const float* W0b_hh = (const float*)d_in[5];
  const float* b0b    = (const float*)d_in[6];
  const float* W1f_ih = (const float*)d_in[7];
  const float* W1f_hh = (const float*)d_in[8];
  const float* b1f    = (const float*)d_in[9];
  const float* W1b_ih = (const float*)d_in[10];
  // d_in[11] (W1b_hh) unused: only the first backward step contributes (h0=c0=0)
  const float* b1b    = (const float*)d_in[12];
  const float* Wl     = (const float*)d_in[13];
  const float* bl     = (const float*)d_in[14];

  const int T = in_sizes[0] / (kB * 4);

  char* ws = (char*)d_ws;
  size_t off = 0;
  auto carve = [&](size_t bytes) {
    char* p = ws + off;
    off = (off + bytes + 255) & ~(size_t)255;
    return p;
  };
  _Float16* h1   = (_Float16*)carve((size_t)kB * T * kH1S * 2);   // 268 MB
  float*    xp1  = (float*)carve((size_t)kB * T * kNG * 4);       // 873 MB
  _Float16* B0   = (_Float16*)carve((size_t)2 * kNG * kKL0 * 2);
  _Float16* B1hh = (_Float16*)carve((size_t)kNG * kKH * 2);
  _Float16* B1if = (_Float16*)carve((size_t)kNG * kKI1 * 2);
  _Float16* B1ib = (_Float16*)carve((size_t)kNG * kKI1 * 2);
  float*    h2f  = (float*)carve((size_t)kB * kH * 4);
  float*    h2b  = (float*)carve((size_t)kB * kH * 4);

  const int packN = 2 * kNG * kKL0 + kNG * kKH + 2 * kNG * kKI1;
  pack_weights<<<(packN + 255) / 256, 256, 0, stream>>>(
      W0f_ih, W0f_hh, W0b_ih, W0b_hh, W1f_hh, W1f_ih, W1b_ih,
      B0, B1hh, B1if, B1ib);
  zero_h1_pad<<<(kB * T + 255) / 256, 256, 0, stream>>>(h1, (size_t)kB * T);
  layer0_scan<<<32, 256, 0, stream>>>(seq, B0, b0f, b0b, h1, T);
  const int Mtiles = kB * T / 16;
  xp1_gemm<<<(Mtiles + 7) / 8, 256, 0, stream>>>(h1, B1if, b1f, xp1, Mtiles);
  layer1f_scan<<<16, 256, 0, stream>>>(B1hh, xp1, h2f, T);
  layer1b_step<<<16, 256, 0, stream>>>(h1, B1ib, b1b, h2b, T);
  head_k<<<1, 256, 0, stream>>>(h2f, h2b, Wl, bl, (float*)d_out);
  (void)n_in; (void)out_size; (void)ws_size;
}
